// LambdaRRDB_3126736192040
// MI455X (gfx1250) — matrix operand order repaired and checked by reading, not compiled, not run on hardware
//
#include <hip/hip_runtime.h>
#include <stdint.h>

typedef __attribute__((ext_vector_type(16))) __bf16    v16bf;
typedef __attribute__((ext_vector_type(8)))  float     v8f;
typedef __attribute__((ext_vector_type(4)))  float     v4f;
typedef __attribute__((ext_vector_type(8)))  uint32_t  v8u32;

#define NPOS 36864
#define HW 192

// native fptrunc f32 -> bf16 (RNE); gfx1250 has hardware bf16 converts
__device__ __forceinline__ __bf16 f2bf(float f) { return (__bf16)f; }
__device__ __forceinline__ uint16_t f2bfu(float f) {
  __bf16 b = (__bf16)f;
  uint16_t u; __builtin_memcpy(&u, &b, 2); return u;
}
__device__ __forceinline__ __bf16 u2bf(uint16_t h) {
  __bf16 b; __builtin_memcpy(&b, &h, 2); return b;
}
__device__ __forceinline__ float bfu2f(uint16_t h) {
  union { uint32_t u; float f; } a; a.u = ((uint32_t)h) << 16; return a.f;
}
// A-matrix 16x32 bf16 per-lane K index for fragment slot e (ISA 7.12.2)
__device__ __forceinline__ int aK(int e, int hi) {
  int j = e >> 1, p = e & 1;
  return ((j & 4) ? 16 : 0) + (hi ? 8 : 0) + 2 * (j & 3) + p;
}
__device__ __forceinline__ v16bf load_afrag(const uint16_t* p, int fragIdx, int lane) {
  // weights pre-swizzled: [frag][lane][16 bf16] -> one 32B per-lane load
  const v8u32 raw = ((const v8u32*)p)[fragIdx * 32 + lane];
  return __builtin_bit_cast(v16bf, raw);
}

// ---------------- weight prep (pre-swizzled into WMMA A-fragment order) ----------------
__global__ void k_prep_qkv(const float* __restrict__ wq, const float* __restrict__ wk,
                           const float* __restrict__ wv, const float* __restrict__ qg,
                           const float* __restrict__ qb, const float* __restrict__ vg,
                           const float* __restrict__ vb, uint16_t* __restrict__ Wb,
                           float* __restrict__ bias) {
  const int i = blockIdx.x * 256 + threadIdx.x;   // [ot(12)][ks(2)][lane(32)][e(16)]
  if (i >= 12 * 2 * 32 * 16) return;
  const int e = i & 15, lane = (i >> 4) & 31, ks = (i >> 9) & 1, ot = i >> 10;
  const int o = ot * 16 + (lane & 15);
  const int c = ks * 32 + aK(e, lane >> 4);
  const float s = rsqrtf(1.0f + 1e-5f);
  float w, sc;
  if (o < 64)       { w = wq[o * 64 + c];         sc = qg[o] * s; }
  else if (o < 128) { w = wk[(o - 64) * 64 + c];  sc = 1.0f; }
  else              { w = wv[(o - 128) * 64 + c]; sc = vg[o - 128] * s; }
  Wb[i] = f2bfu(w * sc);
  if (ks == 0 && e == 0 && lane < 16)
    bias[o] = (o < 64) ? qb[o] : (o < 128 ? 0.0f : vb[o - 128]);
}

__global__ void k_prep_conv(const float* __restrict__ pw, uint16_t* __restrict__ wcvf) {
  // [ky(23)][s(3)][lane(32)][e(16)], K axis padded: kk = u*24 + kx, kx==23 -> 0
  const int i = blockIdx.x * 256 + threadIdx.x;
  if (i >= 23 * 3 * 32 * 16) return;
  const int e = i & 15, lane = (i >> 4) & 31, s = (i >> 9) % 3, ky = i / 1536;
  const int kk = s * 32 + aK(e, lane >> 4);
  const int u = kk / 24, kx = kk - 24 * u;
  const int ko = lane & 15;
  float w = 0.0f;
  if (kx < 23) w = pw[(((size_t)ko * 4 + u) * 23 + ky) * 23 + kx];
  wcvf[i] = f2bfu(w);
}

__global__ void k_prep_cw(const float* __restrict__ cw, uint16_t* __restrict__ cwb) {
  const int i = blockIdx.x * 256 + threadIdx.x;   // [ot(4)][ks(2)][lane(32)][e(16)]
  if (i >= 4 * 2 * 32 * 16) return;
  const int e = i & 15, lane = (i >> 4) & 31, ks = (i >> 9) & 1, ot = i >> 10;
  const int o = ot * 16 + (lane & 15);
  const int c = ks * 32 + aK(e, lane >> 4);
  cwb[i] = f2bfu(cw[o * 64 + c]);
}

__global__ void k_zero(float* __restrict__ p, int nelem) {
  const int i = blockIdx.x * 256 + threadIdx.x;
  if (i < nelem) p[i] = 0.0f;
}

// ---------------- fused q/k/v 1x1 conv (GEMM, WMMA bf16) ----------------
__global__ __launch_bounds__(32) void k_gemm_qkv(const float* __restrict__ xin,
                                                 const uint16_t* __restrict__ Wb,
                                                 const float* __restrict__ bias,
                                                 float* __restrict__ qkv) {
  const int lane = threadIdx.x, hi = lane >> 4, lm = lane & 15;
  const int b = blockIdx.z, ot = blockIdx.y;
  const int col0 = blockIdx.x * 64;
  v8f acc[4] = {};
  for (int ks = 0; ks < 2; ++ks) {
    const v16bf af = load_afrag(Wb, ot * 2 + ks, lane);
#pragma unroll
    for (int t = 0; t < 4; ++t) {
      v16bf bfv;
      const int col = col0 + t * 16 + lm;
      const float* xp = xin + ((size_t)b * 64 + ks * 32 + (hi ? 8 : 0)) * NPOS + col;
#pragma unroll
      for (int e = 0; e < 16; ++e) bfv[e] = f2bf(xp[(size_t)(e + (e & 8)) * NPOS]);
      acc[t] = __builtin_amdgcn_wmma_f32_16x16x32_bf16(false, af, false, bfv,
                                                       (short)0, acc[t], false, false);
    }
  }
#pragma unroll
  for (int t = 0; t < 4; ++t) {
    const int col = col0 + t * 16 + lm;
#pragma unroll
    for (int r = 0; r < 8; ++r) {
      const int o = ot * 16 + r + (hi ? 8 : 0);
      qkv[((size_t)b * 192 + o) * NPOS + col] = acc[t][r] + bias[o];
    }
  }
}

// ---------------- softmax over n for k rows ----------------
__global__ __launch_bounds__(256) void k_softmax(float* __restrict__ qkv) {
  const int row = blockIdx.x;               // b*64 + c
  const int b = row >> 6, c = row & 63;
  float* p = qkv + ((size_t)b * 192 + 64 + c) * NPOS;
  __shared__ float red[256];
  const int tid = threadIdx.x;
  float m = -1e30f;
  for (int i = tid; i < NPOS; i += 256) m = fmaxf(m, p[i]);
  red[tid] = m; __syncthreads();
  for (int s = 128; s > 0; s >>= 1) { if (tid < s) red[tid] = fmaxf(red[tid], red[tid + s]); __syncthreads(); }
  m = red[0]; __syncthreads();
  float sum = 0.0f;
  for (int i = tid; i < NPOS; i += 256) sum += __expf(p[i] - m);
  red[tid] = sum; __syncthreads();
  for (int s = 128; s > 0; s >>= 1) { if (tid < s) red[tid] += red[tid + s]; __syncthreads(); }
  const float inv = 1.0f / red[0]; __syncthreads();
  for (int i = tid; i < NPOS; i += 256) p[i] = __expf(p[i] - m) * inv;
}

// ---------------- content lambda: lam_c[b,kd,v] = sum_{u,n} k*v ----------------
__global__ __launch_bounds__(32) void k_lamc(const float* __restrict__ qkv,
                                             float* __restrict__ lamc) {
  const int lane = threadIdx.x, hi = lane >> 4, lm = lane & 15;
  const int b = blockIdx.z, u = blockIdx.y;
  const int p0 = blockIdx.x * 512;
  const float* kp = qkv + ((size_t)b * 192 + 64 + u * 16) * NPOS + (size_t)lm * NPOS;
  const float* vp = qkv + ((size_t)b * 192 + 128 + u * 16) * NPOS + (size_t)lm * NPOS;
  v8f acc = {};
  for (int pp = p0; pp < p0 + 512; pp += 32) {
    // A: rows of k; per-lane two contiguous 8-float runs -> b128 loads
    const float* ar = kp + pp + hi * 8;
    const v4f a0 = *(const v4f*)(ar);
    const v4f a1 = *(const v4f*)(ar + 4);
    const v4f a2 = *(const v4f*)(ar + 16);
    const v4f a3 = *(const v4f*)(ar + 20);
    // B: columns = v channels; per-lane two contiguous 8-float runs (same k as A)
    const float* br = vp + pp + hi * 8;
    const v4f b0 = *(const v4f*)(br);
    const v4f b1 = *(const v4f*)(br + 4);
    const v4f b2 = *(const v4f*)(br + 16);
    const v4f b3 = *(const v4f*)(br + 20);
    v16bf af, bfv;
#pragma unroll
    for (int j = 0; j < 4; ++j) {
      af[j] = f2bf(a0[j]);  af[4 + j] = f2bf(a1[j]);
      af[8 + j] = f2bf(a2[j]); af[12 + j] = f2bf(a3[j]);
      bfv[j] = f2bf(b0[j]); bfv[4 + j] = f2bf(b1[j]);
      bfv[8 + j] = f2bf(b2[j]); bfv[12 + j] = f2bf(b3[j]);
    }
    acc = __builtin_amdgcn_wmma_f32_16x16x32_bf16(false, af, false, bfv,
                                                  (short)0, acc, false, false);
  }
#pragma unroll
  for (int r = 0; r < 8; ++r)
    atomicAdd(&lamc[b * 256 + (r + (hi ? 8 : 0)) * 16 + lm], acc[r]);
}

// ---------------- position lambda: implicit-GEMM 23x23 conv (WMMA) ----------------
__global__ __launch_bounds__(32) void k_conv(const float* __restrict__ qkv,
                                             const uint16_t* __restrict__ wcvf,
                                             const float* __restrict__ pb,
                                             uint16_t* __restrict__ lamp) {
  const int lane = threadIdx.x, hi = lane >> 4, lm = lane & 15;
  const int img = blockIdx.z, b = img >> 4, v = img & 15;
  const int h = blockIdx.y, w0 = blockIdx.x * 64;
  __shared__ uint32_t lds32[4 * 44];                 // 4 rows x 88 bf16 (stride 88)
  uint16_t* lds16 = (uint16_t*)lds32;
  const float* vbase = qkv + ((size_t)b * 192 + 128 + v) * NPOS;  // + u*16*NPOS
  v8f acc[4] = {};
  for (int ky = 0; ky < 23; ++ky) {
    const int hy = h + ky - 11;
    const bool rowok = (hy >= 0) & (hy < HW);
    __syncthreads();
    for (int c = lane; c < 88; c += 32) {            // 4-bf16 packed chunks
      const int u = c / 22, q4 = (c % 22) * 4;
      const float* rowp = vbase + (size_t)u * 16 * NPOS + (size_t)(rowok ? hy : 0) * HW;
      uint32_t w01 = 0, w23 = 0;
      if (rowok) {
        const int wbase = w0 - 11 + q4;
#pragma unroll
        for (int j = 0; j < 2; ++j) {
          const int wa = wbase + j;
          if (wa >= 0 && wa < HW) w01 |= (uint32_t)f2bfu(rowp[wa]) << (16 * j);
        }
#pragma unroll
        for (int j = 0; j < 2; ++j) {
          const int wa = wbase + 2 + j;
          if (wa >= 0 && wa < HW) w23 |= (uint32_t)f2bfu(rowp[wa]) << (16 * j);
        }
      }
      lds32[u * 44 + (q4 >> 1)] = w01;
      lds32[u * 44 + (q4 >> 1) + 1] = w23;
    }
    __syncthreads();
#pragma unroll
    for (int s = 0; s < 3; ++s) {
      const v16bf af = load_afrag(wcvf, ky * 3 + s, lane);
#pragma unroll
      for (int t = 0; t < 4; ++t) {
        v16bf bfv;
#pragma unroll
        for (int r2 = 0; r2 < 2; ++r2) {
          // K padded as kk = u*24 + kx : every aligned 8-run sits inside one u
          const int klo = s * 32 + r2 * 16;
          const int khi = klo + 8;
          const int clo = (klo / 24) * 88 + (klo % 24);
          const int chi = (khi / 24) * 88 + (khi % 24);
          const int basei = (hi ? chi : clo) + t * 16 + lm;
#pragma unroll
          for (int j = 0; j < 8; ++j) bfv[r2 * 8 + j] = u2bf(lds16[basei + j]);
        }
        acc[t] = __builtin_amdgcn_wmma_f32_16x16x32_bf16(false, af, false, bfv,
                                                         (short)0, acc[t], false, false);
      }
    }
  }
#pragma unroll
  for (int t = 0; t < 4; ++t)
#pragma unroll
    for (int r = 0; r < 8; ++r) {
      const int kd = r + (hi ? 8 : 0);
      const int w = w0 + t * 16 + lm;
      lamp[(((size_t)b * 16 + kd) * 16 + v) * NPOS + h * HW + w] = f2bfu(acc[t][r] + pb[kd]);
    }
}

// ---------------- y = q . (lam_c + lam_p), per pixel ----------------
__global__ __launch_bounds__(256) void k_combine(const float* __restrict__ qkv,
                                                 const float* __restrict__ lamc,
                                                 const uint16_t* __restrict__ lamp,
                                                 float* __restrict__ y) {
  __shared__ float lc[256];
  const int tid = threadIdx.x;
  const size_t gid = (size_t)blockIdx.x * 256 + tid;
  const int b = (int)(gid / NPOS);
  const int pos = (int)(gid % NPOS);
  lc[tid] = lamc[b * 256 + tid];   // 36864 % 256 == 0 -> b uniform per block
  __syncthreads();
  const float* qb = qkv + (size_t)b * 192 * NPOS + pos;
  const uint16_t* lp = lamp + (size_t)b * 256 * NPOS + pos;
  float acc[64] = {};
#pragma unroll
  for (int kd = 0; kd < 16; ++kd) {
    float qv[4];
#pragma unroll
    for (int hh = 0; hh < 4; ++hh) qv[hh] = qb[(size_t)(hh * 16 + kd) * NPOS];
#pragma unroll
    for (int vv = 0; vv < 16; ++vv) {
      const float l = bfu2f(lp[(size_t)(kd * 16 + vv) * NPOS]) + lc[kd * 16 + vv];
#pragma unroll
      for (int hh = 0; hh < 4; ++hh) acc[hh * 16 + vv] += qv[hh] * l;
    }
  }
#pragma unroll
  for (int o = 0; o < 64; ++o)
    y[((size_t)b * 64 + o) * NPOS + pos] = acc[o];
}

// ---------------- GroupNorm(8 groups) ----------------
__global__ __launch_bounds__(256) void k_gnorm(const float* __restrict__ y,
                                               const float* __restrict__ gg,
                                               const float* __restrict__ gb,
                                               float* __restrict__ o) {
  const int b = blockIdx.x >> 3, g = blockIdx.x & 7;
  const float* base = y + ((size_t)b * 64 + g * 8) * NPOS;
  float* obase = o + ((size_t)b * 64 + g * 8) * NPOS;
  const int tid = threadIdx.x;
  __shared__ float rs[256], rq[256];
  float s = 0.0f, q = 0.0f;
  for (int i = tid; i < 8 * NPOS; i += 256) { const float v = base[i]; s += v; q += v * v; }
  rs[tid] = s; rq[tid] = q; __syncthreads();
  for (int k = 128; k > 0; k >>= 1) {
    if (tid < k) { rs[tid] += rs[tid + k]; rq[tid] += rq[tid + k]; }
    __syncthreads();
  }
  const float cnt = 8.0f * NPOS;
  const float mean = rs[0] / cnt;
  const float var = rq[0] / cnt - mean * mean;
  const float inv = rsqrtf(var + 1e-5f);
  for (int i = tid; i < 8 * NPOS; i += 256) {
    const int c = g * 8 + i / NPOS;
    obase[i] = (base[i] - mean) * inv * gg[c] + gb[c];
  }
}

// ---------------- final 1x1 conv + 0.2*(.)+x residual (WMMA) ----------------
__global__ __launch_bounds__(32) void k_final(const float* __restrict__ feat,
                                              const uint16_t* __restrict__ cwb,
                                              const float* __restrict__ cb,
                                              const float* __restrict__ x0,
                                              float* __restrict__ out) {
  const int lane = threadIdx.x, hi = lane >> 4, lm = lane & 15;
  const int b = blockIdx.z, ot = blockIdx.y;
  const int col0 = blockIdx.x * 64;
  v8f acc[4] = {};
  for (int ks = 0; ks < 2; ++ks) {
    const v16bf af = load_afrag(cwb, ot * 2 + ks, lane);
#pragma unroll
    for (int t = 0; t < 4; ++t) {
      v16bf bfv;
      const int col = col0 + t * 16 + lm;
      const float* xp = feat + ((size_t)b * 64 + ks * 32 + (hi ? 8 : 0)) * NPOS + col;
#pragma unroll
      for (int e = 0; e < 16; ++e) bfv[e] = f2bf(xp[(size_t)(e + (e & 8)) * NPOS]);
      acc[t] = __builtin_amdgcn_wmma_f32_16x16x32_bf16(false, af, false, bfv,
                                                       (short)0, acc[t], false, false);
    }
  }
#pragma unroll
  for (int t = 0; t < 4; ++t)
#pragma unroll
    for (int r = 0; r < 8; ++r) {
      const int o = ot * 16 + r + (hi ? 8 : 0);
      const size_t idx = ((size_t)b * 64 + o) * NPOS + col0 + t * 16 + lm;
      out[idx] = 0.2f * (acc[t][r] + cb[o]) + x0[idx];
    }
}

extern "C" void kernel_launch(void* const* d_in, const int* in_sizes, int n_in,
                              void* d_out, int out_size, void* d_ws, size_t ws_size,
                              hipStream_t stream) {
  (void)in_sizes; (void)n_in; (void)out_size; (void)ws_size;
  const float* x = (const float*)d_in[0];
  char* ws = (char*)d_ws;
  size_t off = 0;
  auto alloc = [&](size_t bytes) -> void* {
    void* p = ws + off;
    off = (off + bytes + 255) & ~(size_t)255;
    return p;
  };
  float*    qkv  = (float*)alloc((size_t)4 * 192 * NPOS * 4);
  uint16_t* lamp = (uint16_t*)alloc((size_t)4 * 256 * NPOS * 2);
  float*    ybuf = (float*)alloc((size_t)4 * 64 * NPOS * 4);
  float*    feat = (float*)alloc((size_t)4 * 64 * NPOS * 4);
  float*    lamc = (float*)alloc(4 * 256 * 4);
  uint16_t* Wb[3]; float* bias[3]; uint16_t* wcv[3];
  for (int l = 0; l < 3; ++l) {
    Wb[l]   = (uint16_t*)alloc(12 * 2 * 32 * 16 * 2);
    bias[l] = (float*)alloc(192 * 4);
    wcv[l]  = (uint16_t*)alloc(23 * 3 * 32 * 16 * 2);
  }
  uint16_t* cwb = (uint16_t*)alloc(4 * 2 * 32 * 16 * 2);

  for (int l = 0; l < 3; ++l) {
    const int base = 1 + l * 9;
    k_prep_qkv<<<(12 * 2 * 32 * 16 + 255) / 256, 256, 0, stream>>>(
        (const float*)d_in[base + 0], (const float*)d_in[base + 1], (const float*)d_in[base + 2],
        (const float*)d_in[base + 3], (const float*)d_in[base + 4], (const float*)d_in[base + 5],
        (const float*)d_in[base + 6], Wb[l], bias[l]);
    k_prep_conv<<<(23 * 3 * 32 * 16 + 255) / 256, 256, 0, stream>>>((const float*)d_in[base + 7], wcv[l]);
  }
  k_prep_cw<<<(4 * 2 * 32 * 16 + 255) / 256, 256, 0, stream>>>((const float*)d_in[32], cwb);

  const float* gn1g = (const float*)d_in[28];
  const float* gn1b = (const float*)d_in[29];
  const float* gn3g = (const float*)d_in[30];
  const float* gn3b = (const float*)d_in[31];

  const float* cur = x;
  for (int l = 0; l < 3; ++l) {
    const int base = 1 + l * 9;
    k_gemm_qkv<<<dim3(576, 12, 4), 32, 0, stream>>>(cur, Wb[l], bias[l], qkv);
    k_softmax<<<256, 256, 0, stream>>>(qkv);
    k_zero<<<4, 256, 0, stream>>>(lamc, 1024);
    k_lamc<<<dim3(72, 4, 4), 32, 0, stream>>>(qkv, lamc);
    k_conv<<<dim3(3, 192, 64), 32, 0, stream>>>(qkv, wcv[l], (const float*)d_in[base + 8], lamp);
    k_combine<<<(4 * NPOS) / 256, 256, 0, stream>>>(qkv, lamc, lamp, ybuf);
    k_gnorm<<<32, 256, 0, stream>>>(ybuf, (l == 2) ? gn3g : gn1g, (l == 2) ? gn3b : gn1b, feat);
    cur = feat;
  }
  k_final<<<dim3(576, 4, 4), 32, 0, stream>>>(feat, cwb, (const float*)d_in[33], x, (float*)d_out);
}
